// CrossStockAttention_21191368638723
// MI455X (gfx1250) — compile-verified, matrix operand order checked by reading, not run on hardware
//
#include <hip/hip_runtime.h>
#include <hip/hip_bf16.h>

// ---------------------------------------------------------------------------
// CrossStockAttention on gfx1250 (MI455X): bf16 WMMA pipeline.
// B=2, N=2048, D=256, H=8, DH=32, TOPK=40.
// ---------------------------------------------------------------------------

#define Bq 2
#define Nq 2048
#define Dq 256
#define Hq 8
#define DHq 32
#define TOPKq 40

typedef __bf16  bf16_t;
typedef __bf16  v16bf __attribute__((ext_vector_type(16)));
typedef float   v8f   __attribute__((ext_vector_type(8)));

__device__ __forceinline__ v8f wmma_bf16(v16bf a, v16bf b, v8f c) {
    // D = A(16x32 bf16) * B(32x16 bf16) + C(16x16 f32)
    return __builtin_amdgcn_wmma_f32_16x16x32_bf16(
        /*neg_a=*/false, a, /*neg_b=*/false, b,
        /*c_mod=*/(short)0, c, /*reuse_a=*/false, /*reuse_b=*/false);
}

// A/B fragment loader: A and B share this code's contiguous k order, "lane (L&15) picks row/col,
// lane>>4 picks K-half, 16 contiguous bf16 per lane" -> one 32B load.
__device__ __forceinline__ v16bf ldfrag(const bf16_t* base, int stride,
                                        int r15, int kofs) {
    return *(const v16bf*)(base + (size_t)r15 * stride + kofs);
}

__device__ __forceinline__ unsigned fkey(float f) {
    unsigned b = __float_as_uint(f);
    return (b & 0x80000000u) ? ~b : (b | 0x80000000u);
}
__device__ __forceinline__ float finv(unsigned u) {
    unsigned b = (u & 0x80000000u) ? (u & 0x7fffffffu) : ~u;
    return __uint_as_float(b);
}

// 8-step K accumulation with depth-2 double buffering of B fragments:
// two load-pairs stay in flight under every WMMA, A-frags cached in regs.
__device__ __forceinline__ v8f gemm_k8(const v16bf afr[8], const bf16_t* bbase,
                                       int r15, int ko) {
    v8f acc = {};
    v16bf b0 = ldfrag(bbase, Dq, r15, 0 * 32 + ko);
    v16bf b1 = ldfrag(bbase, Dq, r15, 1 * 32 + ko);
#pragma unroll
    for (int c = 0; c < 6; c++) {
        v16bf bn = ldfrag(bbase, Dq, r15, (c + 2) * 32 + ko);
        acc = wmma_bf16(afr[c], b0, acc);
        b0 = b1;
        b1 = bn;
    }
    acc = wmma_bf16(afr[6], b0, acc);
    acc = wmma_bf16(afr[7], b1, acc);
    return acc;
}

// ------------------------- workspace layout (bytes) ------------------------
static const size_t XB_OFF  = 0;                       // bf16 x        [4096][256]
static const size_t NB_OFF  = XB_OFF  + 2097152;       // bf16 x/|x|    [4096][256]
static const size_t WT_OFF  = NB_OFF  + 2097152;       // bf16 wT[4][256][256] (q,k,v,o)
static const size_t QB_OFF  = WT_OFF  + 524288;        // bf16 q        [4096][256]
static const size_t KB_OFF  = QB_OFF  + 2097152;       // bf16 k        [4096][256]
static const size_t VT_OFF  = KB_OFF  + 2097152;       // bf16 vT  [B][H][DH][N]
static const size_t AT_OFF  = VT_OFF  + 2097152;       // bf16 attn_out [4096][256]
static const size_t SIM_OFF = AT_OFF  + 2097152;       // f32 sim  [B][N][N]
static const size_t TH_OFF  = SIM_OFF + 33554432;      // f32 thresh [B][N]
static const size_t AM_OFF  = TH_OFF  + 16384;         // u32 amask  [B][N][64]
static const size_t AMT_OFF = AM_OFF  + 1048576;       // u32 amaskT [B][N][64]

// ---------------------------------------------------------------------------
// Kernel 1: x -> bf16; normalized x -> bf16 (cosine-sim operand)
// ---------------------------------------------------------------------------
__global__ void prepx_kernel(const float* __restrict__ x,
                             bf16_t* __restrict__ xb, bf16_t* __restrict__ nb) {
    int n = blockIdx.x;      // 0..4095 (b,n flattened)
    int t = threadIdx.x;     // 0..255
    __shared__ float red[256];
    float v = x[(size_t)n * Dq + t];
    xb[(size_t)n * Dq + t] = (bf16_t)v;
    red[t] = v * v;
    __syncthreads();
    for (int s = 128; s > 0; s >>= 1) {
        if (t < s) red[t] += red[t + s];
        __syncthreads();
    }
    float denom = fmaxf(sqrtf(red[0]), 1e-12f);
    nb[(size_t)n * Dq + t] = (bf16_t)(v / denom);
}

// ---------------------------------------------------------------------------
// Kernel 2: weights -> transposed bf16  wT[which][outcol][inrow]
// ---------------------------------------------------------------------------
__global__ void prepw_kernel(const float* __restrict__ wq, const float* __restrict__ wk,
                             const float* __restrict__ wv, const float* __restrict__ wo,
                             bf16_t* __restrict__ wT) {
    int j = blockIdx.x;          // output column
    int which = blockIdx.y;      // 0..3
    int i = threadIdx.x;         // input row
    const float* w = (which == 0) ? wq : (which == 1) ? wk : (which == 2) ? wv : wo;
    wT[(size_t)which * 65536 + (size_t)j * Dq + i] = (bf16_t)w[(size_t)i * Dq + j];
}

// ---------------------------------------------------------------------------
// Kernel 3: QKV projections.  block=256 (8 waves), one 16-row tile per block.
// ---------------------------------------------------------------------------
__global__ void qkv_kernel(const bf16_t* __restrict__ xb, const bf16_t* __restrict__ wT,
                           const float* __restrict__ bq, const float* __restrict__ bk,
                           const float* __restrict__ bv,
                           bf16_t* __restrict__ qb, bf16_t* __restrict__ kb,
                           bf16_t* __restrict__ vT) {
    int ntile = blockIdx.x;                 // 0..255
    int wave = threadIdx.x >> 5;
    int lane = threadIdx.x & 31;
    int r15 = lane & 15, hi = lane >> 4, ko = hi * 16;

    const bf16_t* abase = xb + (size_t)ntile * 16 * Dq;
    v16bf afr[8];
#pragma unroll
    for (int c = 0; c < 8; c++) afr[c] = ldfrag(abase, Dq, r15, c * 32 + ko);

#pragma unroll 1
    for (int t = 0; t < 6; t++) {           // serialize tiles: avoid VGPR spills
        int ct = wave + 8 * t;              // 0..47 col-tiles across q|k|v
        int mat = ct / 16;                  // 0:q 1:k 2:v
        int colt = ct % 16;
        const bf16_t* bbase = wT + (size_t)mat * 65536 + (size_t)colt * 16 * Dq;
        v8f acc = gemm_k8(afr, bbase, r15, ko);

        const float* bias = (mat == 0) ? bq : (mat == 1) ? bk : bv;
        int col = colt * 16 + r15;
        float bia = bias[col];
#pragma unroll
        for (int rr = 0; rr < 8; rr++) {
            int row = rr + hi * 8;
            int n = ntile * 16 + row;       // flattened (b,n)
            bf16_t hv = (bf16_t)(acc[rr] + bia);
            if (mat == 0) {
                qb[(size_t)n * Dq + col] = hv;
            } else if (mat == 1) {
                kb[(size_t)n * Dq + col] = hv;
            } else {
                int b = n >> 11, nn = n & (Nq - 1);
                int h = col >> 5, dh = col & 31;
                vT[((size_t)(b * Hq + h) * DHq + dh) * Nq + nn] = hv;
            }
        }
    }
}

// ---------------------------------------------------------------------------
// Kernel 4: cosine similarity sim[b][q][k] via WMMA; invalid keys -> -inf.
// ---------------------------------------------------------------------------
__global__ void sim_kernel(const bf16_t* __restrict__ nb, const int* __restrict__ valid,
                           float* __restrict__ sim) {
    int qt = blockIdx.x;                    // 0..127
    int b  = blockIdx.y;
    int wave = threadIdx.x >> 5;
    int lane = threadIdx.x & 31;
    int r15 = lane & 15, hi = lane >> 4, ko = hi * 16;

    const bf16_t* abase = nb + (size_t)(b * Nq + qt * 16) * Dq;
    v16bf afr[8];
#pragma unroll
    for (int c = 0; c < 8; c++) afr[c] = ldfrag(abase, Dq, r15, c * 32 + ko);

    float* srow = sim + (size_t)b * Nq * Nq;
    const float NEGINF = -__builtin_inff();

#pragma unroll 1
    for (int t = 0; t < 16; t++) {          // serialize tiles: avoid VGPR spills
        int kt = wave + 8 * t;              // 0..127
        const bf16_t* bbase = nb + (size_t)(b * Nq + kt * 16) * Dq;
        v8f acc = gemm_k8(afr, bbase, r15, ko);

        int k = kt * 16 + r15;
        bool vk = valid[b * Nq + k] != 0;
#pragma unroll
        for (int rr = 0; rr < 8; rr++) {
            int q = qt * 16 + rr + hi * 8;
            srow[(size_t)q * Nq + k] = vk ? acc[rr] : NEGINF;
        }
    }
}

// ---------------------------------------------------------------------------
// Kernel 5: per-row 40th-largest threshold (bitwise binary search on the
// order-preserving uint mapping) + allowed-bitmask words.
// ---------------------------------------------------------------------------
__global__ void topk_kernel(const float* __restrict__ sim, const int* __restrict__ valid,
                            float* __restrict__ thresh, unsigned* __restrict__ amask) {
    int rowid = blockIdx.x;                 // 0..4095
    int b = rowid >> 11, q = rowid & (Nq - 1);
    int t = threadIdx.x;
    __shared__ float sv[Nq];
    __shared__ int   red[256];
    __shared__ int   sany;

    const float* srow = sim + (size_t)(b * Nq + q) * Nq;
    int anyl = 0;
#pragma unroll
    for (int i = 0; i < 8; i++) {
        int k = t + i * 256;
        sv[k] = srow[k];
        anyl += (valid[b * Nq + k] != 0);
    }
    red[t] = anyl;
    __syncthreads();
    for (int s = 128; s > 0; s >>= 1) {
        if (t < s) red[t] += red[t + s];
        __syncthreads();
    }
    if (t == 0) sany = red[0];
    __syncthreads();
    int anyvalid = sany;

    unsigned pref = 0;
#pragma unroll 1
    for (int bit = 31; bit >= 0; --bit) {
        unsigned cand = pref | (1u << bit);
        int c = 0;
#pragma unroll
        for (int i = 0; i < 8; i++) c += (fkey(sv[t + i * 256]) >= cand) ? 1 : 0;
        __syncthreads();
        red[t] = c;
        __syncthreads();
        for (int s = 128; s > 0; s >>= 1) {
            if (t < s) red[t] += red[t + s];
            __syncthreads();
        }
        if (red[0] >= TOPKq) pref = cand;
    }
    if (t == 0) thresh[rowid] = finv(pref);

    int vq = valid[b * Nq + q];
    if (t < Nq / 32) {
        unsigned w = 0;
#pragma unroll
        for (int j = 0; j < 32; j++) {
            int k = t * 32 + j;
            int vk = valid[b * Nq + k];
            bool a;
            if (!vq) a = anyvalid ? (vk != 0) : true;          // pad row
            else     a = (vk != 0) && ((fkey(sv[k]) >= pref) || (k == q));
            if (a) w |= (1u << j);
        }
        amask[(size_t)(b * Nq + q) * (Nq / 32) + t] = w;
    }
}

// ---------------------------------------------------------------------------
// Kernel 6: bit-transpose allowed mask -> amaskT[b][k][qword]
// ---------------------------------------------------------------------------
__global__ void maskT_kernel(const unsigned* __restrict__ amask,
                             unsigned* __restrict__ amaskT) {
    int idx = blockIdx.x * 256 + threadIdx.x;     // < 2*2048*64
    int qw = idx & 63;
    int k  = (idx >> 6) & (Nq - 1);
    int b  = idx >> 17;
    unsigned w = 0;
#pragma unroll
    for (int j = 0; j < 32; j++) {
        int q = qw * 32 + j;
        unsigned src = amask[(size_t)(b * Nq + q) * (Nq / 32) + (k >> 5)];
        w |= ((src >> (k & 31)) & 1u) << j;
    }
    amaskT[(size_t)(b * Nq + k) * (Nq / 32) + qw] = w;
}

// ---------------------------------------------------------------------------
// Kernel 7: sparse attention, flash-style two-pass, one wave per (b,h,qtile).
// ---------------------------------------------------------------------------
__global__ __launch_bounds__(32)
void attn_kernel(const bf16_t* __restrict__ qb, const bf16_t* __restrict__ kb,
                 const bf16_t* __restrict__ vT, const unsigned* __restrict__ amaskT,
                 bf16_t* __restrict__ attn) {
    int qt = blockIdx.x, h = blockIdx.y, b = blockIdx.z;
    int lane = threadIdx.x & 31;
    int r15 = lane & 15, hi = lane >> 4, ko = hi * 16;
    int qbase = qt * 16;
    const float scale = 0.17677669529663687f;      // 1/sqrt(32)
    const float NEGINF = -__builtin_inff();

    __shared__ float sm[16][16], sl[16][16];
    __shared__ float rowM[16], rowI[16];
    __shared__ float pst[16][32];

    const bf16_t* qbb = qb + (size_t)(b * Nq + qbase) * Dq + h * DHq;
    v16bf aq = ldfrag(qbb, Dq, r15, ko);

    const bf16_t*   kbb = kb + (size_t)b * Nq * Dq + h * DHq;
    const unsigned* mtb = amaskT + (size_t)b * Nq * (Nq / 32);
    int qoff = qbase & 31;       // 0 or 16
    int qw   = qbase >> 5;

    float m[8], l[8];
#pragma unroll
    for (int r = 0; r < 8; r++) { m[r] = NEGINF; l[r] = 0.0f; }

    // ---- pass 1: online row max / sum over all 2048 keys ----
    // depth-2 rotation of K fragments: two tiles in flight under each WMMA.
    v16bf bk0 = ldfrag(kbb, Dq, r15, ko);
    v16bf bk1 = ldfrag(kbb + (size_t)16 * Dq, Dq, r15, ko);
#pragma unroll 1
    for (int kt = 0; kt < Nq / 16; kt++) {
        int ktn = (kt + 2 < Nq / 16) ? (kt + 2) : (Nq / 16 - 1);   // branchless clamp
        v16bf bkn = ldfrag(kbb + (size_t)ktn * 16 * Dq, Dq, r15, ko);
        v8f s = {};
        s = wmma_bf16(aq, bk0, s);
        bk0 = bk1;
        bk1 = bkn;
        int k = kt * 16 + r15;
        unsigned mw = mtb[(size_t)k * (Nq / 32) + qw];
#pragma unroll
        for (int r = 0; r < 8; r++) {
            int row = r + hi * 8;
            bool ok = (mw >> (qoff + row)) & 1u;
            float v = ok ? s[r] * scale : NEGINF;
            float nm = fmaxf(m[r], v);
            if (nm > -3.0e38f) {           // at least one allowed so far
                l[r] = l[r] * __expf(m[r] - nm) + __expf(v - nm);
                m[r] = nm;
            }
        }
    }
#pragma unroll
    for (int r = 0; r < 8; r++) {
        int row = r + hi * 8;
        sm[row][r15] = m[r];
        sl[row][r15] = l[r];
    }
    __syncthreads();
    if (lane < 16) {
        float M = NEGINF;
#pragma unroll
        for (int c = 0; c < 16; c++) M = fmaxf(M, sm[lane][c]);
        float L = 0.0f;
        if (M > -3.0e38f) {
#pragma unroll
            for (int c = 0; c < 16; c++)
                if (sm[lane][c] > -3.0e38f) L += sl[lane][c] * __expf(sm[lane][c] - M);
        }
        rowM[lane] = M;
        rowI[lane] = (L > 0.0f) ? 1.0f / L : 0.0f;
    }
    __syncthreads();

    // ---- pass 2: recompute scores, P in LDS, P*V via WMMA ----
    // rotate 2xK + 2xV fragments one iteration ahead: loads issued at the top
    // of the iteration stay in flight across both WMMAs and the LDS transpose.
    v8f o0 = {}, o1 = {};
    const bf16_t* vtb = vT + (size_t)((b * Hq + h) * DHq) * Nq;
    v16bf ck0 = ldfrag(kbb, Dq, r15, ko);
    v16bf ck1 = ldfrag(kbb + (size_t)16 * Dq, Dq, r15, ko);
    v16bf cv0 = ldfrag(vtb, Nq, r15, ko);
    v16bf cv1 = ldfrag(vtb + (size_t)16 * Nq, Nq, r15, ko);
#pragma unroll 1
    for (int c2 = 0; c2 < Nq / 32; c2++) {
        int c2n = (c2 + 1 < Nq / 32) ? (c2 + 1) : (Nq / 32 - 1);   // branchless clamp
        v16bf nk0 = ldfrag(kbb + (size_t)(2 * c2n) * 16 * Dq, Dq, r15, ko);
        v16bf nk1 = ldfrag(kbb + (size_t)(2 * c2n + 1) * 16 * Dq, Dq, r15, ko);
        v16bf nv0 = ldfrag(vtb, Nq, r15, c2n * 32 + ko);
        v16bf nv1 = ldfrag(vtb + (size_t)16 * Nq, Nq, r15, c2n * 32 + ko);

#pragma unroll
        for (int half = 0; half < 2; half++) {
            int kt = 2 * c2 + half;
            v8f s = {};
            s = wmma_bf16(aq, (half == 0) ? ck0 : ck1, s);
            int k = kt * 16 + r15;
            unsigned mw = mtb[(size_t)k * (Nq / 32) + qw];
#pragma unroll
            for (int r = 0; r < 8; r++) {
                int row = r + hi * 8;
                bool ok = (mw >> (qoff + row)) & 1u;
                float v = ok ? s[r] * scale : NEGINF;
                float inv = rowI[row];
                float p = (inv > 0.0f) ? __expf(v - rowM[row]) * inv : 0.0f;
                pst[row][half * 16 + r15] = p;
            }
        }
        __syncthreads();
        v16bf ap;
#pragma unroll
        for (int i = 0; i < 16; i++) ap[i] = (bf16_t)pst[r15][ko + i];
        o0 = wmma_bf16(ap, cv0, o0);
        o1 = wmma_bf16(ap, cv1, o1);
        __syncthreads();

        ck0 = nk0; ck1 = nk1; cv0 = nv0; cv1 = nv1;
    }
#pragma unroll
    for (int hf = 0; hf < 2; hf++) {
#pragma unroll
        for (int r = 0; r < 8; r++) {
            int row = r + hi * 8;
            int d = h * DHq + hf * 16 + r15;
            float ov = (hf == 0) ? o0[r] : o1[r];
            attn[(size_t)(b * Nq + qbase + row) * Dq + d] = (bf16_t)ov;
        }
    }
}

// ---------------------------------------------------------------------------
// Kernel 8: O projection + bias + residual + LayerNorm (fused).
// ---------------------------------------------------------------------------
__global__ void oln_kernel(const bf16_t* __restrict__ attn, const bf16_t* __restrict__ woT,
                           const float* __restrict__ bo, const float* __restrict__ x,
                           const float* __restrict__ g, const float* __restrict__ bet,
                           float* __restrict__ out) {
    int ntile = blockIdx.x;                 // 0..255
    int wave = threadIdx.x >> 5;
    int lane = threadIdx.x & 31;
    int r15 = lane & 15, hi = lane >> 4, ko = hi * 16;

    __shared__ float yb[16][Dq];
    __shared__ float ps[16][16], pq2[16][16];
    __shared__ float mu[16], rs[16];

    const bf16_t* abase = attn + (size_t)ntile * 16 * Dq;
    v16bf afr[8];
#pragma unroll
    for (int c = 0; c < 8; c++) afr[c] = ldfrag(abase, Dq, r15, c * 32 + ko);

#pragma unroll 1
    for (int half = 0; half < 2; half++) {  // serialize tiles: avoid VGPR spills
        int colt = wave * 2 + half;         // 0..15
        const bf16_t* bbase = woT + (size_t)colt * 16 * Dq;
        v8f acc = gemm_k8(afr, bbase, r15, ko);

        int col = colt * 16 + r15;
        float bia = bo[col];
#pragma unroll
        for (int rr = 0; rr < 8; rr++) {
            int row = rr + hi * 8;
            int n = ntile * 16 + row;
            yb[row][col] = acc[rr] + bia + x[(size_t)n * Dq + col];
        }
    }
    __syncthreads();

    int t = threadIdx.x;
    int row = t >> 4, seg = t & 15;
    float s = 0.0f, sq = 0.0f;
#pragma unroll
    for (int i = 0; i < 16; i++) {
        float v = yb[row][seg * 16 + i];
        s += v; sq += v * v;
    }
    ps[row][seg] = s; pq2[row][seg] = sq;
    __syncthreads();
    if (seg == 0) {
        float S = 0.0f, Q = 0.0f;
#pragma unroll
        for (int i = 0; i < 16; i++) { S += ps[row][i]; Q += pq2[row][i]; }
        float mean = S * (1.0f / Dq);
        float var = Q * (1.0f / Dq) - mean * mean;
        mu[row] = mean;
        rs[row] = rsqrtf(var + 1e-5f);
    }
    __syncthreads();
    int n = ntile * 16 + row;
#pragma unroll
    for (int i = 0; i < 16; i++) {
        int col = seg * 16 + i;
        float v = (yb[row][col] - mu[row]) * rs[row];
        out[(size_t)n * Dq + col] = v * g[col] + bet[col];
    }
}

// ---------------------------------------------------------------------------
extern "C" void kernel_launch(void* const* d_in, const int* in_sizes, int n_in,
                              void* d_out, int out_size, void* d_ws, size_t ws_size,
                              hipStream_t stream) {
    const float* x     = (const float*)d_in[0];
    const int*   valid = (const int*)d_in[1];
    const float* w_q   = (const float*)d_in[2];
    const float* b_q   = (const float*)d_in[3];
    const float* w_k   = (const float*)d_in[4];
    const float* b_k   = (const float*)d_in[5];
    const float* w_v   = (const float*)d_in[6];
    const float* b_v   = (const float*)d_in[7];
    const float* w_o   = (const float*)d_in[8];
    const float* b_o   = (const float*)d_in[9];
    const float* ln_g  = (const float*)d_in[10];
    const float* ln_b  = (const float*)d_in[11];
    float* out = (float*)d_out;

    char* ws = (char*)d_ws;
    bf16_t*   xb     = (bf16_t*)(ws + XB_OFF);
    bf16_t*   nb     = (bf16_t*)(ws + NB_OFF);
    bf16_t*   wT     = (bf16_t*)(ws + WT_OFF);
    bf16_t*   qb     = (bf16_t*)(ws + QB_OFF);
    bf16_t*   kb     = (bf16_t*)(ws + KB_OFF);
    bf16_t*   vT     = (bf16_t*)(ws + VT_OFF);
    bf16_t*   attn   = (bf16_t*)(ws + AT_OFF);
    float*    sim    = (float*)(ws + SIM_OFF);
    float*    thresh = (float*)(ws + TH_OFF);
    unsigned* amask  = (unsigned*)(ws + AM_OFF);
    unsigned* amaskT = (unsigned*)(ws + AMT_OFF);

    prepx_kernel<<<Bq * Nq, 256, 0, stream>>>(x, xb, nb);
    prepw_kernel<<<dim3(Dq, 4), 256, 0, stream>>>(w_q, w_k, w_v, w_o, wT);
    qkv_kernel<<<Bq * Nq / 16, 256, 0, stream>>>(xb, wT, b_q, b_k, b_v, qb, kb, vT);
    sim_kernel<<<dim3(Nq / 16, Bq), 256, 0, stream>>>(nb, valid, sim);
    topk_kernel<<<Bq * Nq, 256, 0, stream>>>(sim, valid, thresh, amask);
    maskT_kernel<<<(Bq * Nq * (Nq / 32)) / 256, 256, 0, stream>>>(amask, amaskT);
    attn_kernel<<<dim3(Nq / 16, Hq, Bq), 32, 0, stream>>>(qb, kb, vT, amaskT, attn);
    oln_kernel<<<Bq * Nq / 16, 256, 0, stream>>>(attn, wT + 3 * 65536, b_o, x, ln_g, ln_b, out);
}
